// AgentAttention_58110907515003
// MI455X (gfx1250) — matrix operand order repaired and checked by reading, not compiled, not run on hardware
//
#include <hip/hip_runtime.h>
#include <hip/hip_bf16.h>

typedef __attribute__((ext_vector_type(16))) __bf16 v16bf;
typedef __attribute__((ext_vector_type(8)))  __bf16 v8bf;
typedef __attribute__((ext_vector_type(4)))  __bf16 v4bf;
typedef __attribute__((ext_vector_type(8)))  float  v8f;
typedef __attribute__((ext_vector_type(4)))  unsigned int u32x4;
typedef __attribute__((ext_vector_type(2)))  unsigned int u32x2;

constexpr int BATCH = 32;
constexpr int CDIM  = 384;
constexpr int HEADS = 12;
constexpr int HD    = 32;
constexpr int HW    = 56;
constexpr int NPOS  = HW * HW;   // 3136
constexpr int AG    = 49;
constexpr int N3C   = 3 * CDIM;  // 1152

// ---------- vectorized helpers ----------
__device__ inline v8bf ld8(const __bf16* p) {            // 16-byte load
    u32x4 t = *(const u32x4*)(const void*)p;
    return __builtin_bit_cast(v8bf, t);
}
__device__ inline v4bf ld4(const __bf16* p) {            // 8-byte load
    u32x2 t = *(const u32x2*)(const void*)p;
    return __builtin_bit_cast(v4bf, t);
}
__device__ inline v16bf combine(v8bf lo, v8bf hi) {
    return __builtin_shufflevector(lo, hi, 0, 1, 2, 3, 4, 5, 6, 7,
                                   8, 9, 10, 11, 12, 13, 14, 15);
}
__device__ inline unsigned pk2f(float a, float b) {      // packed bf16 pair
    union { __bf16 h[2]; unsigned u; } t;
    t.h[0] = (__bf16)a; t.h[1] = (__bf16)b;
    return t.u;
}
__device__ inline unsigned pk2b(__bf16 a, __bf16 b) {
    union { __bf16 h[2]; unsigned u; } t;
    t.h[0] = a; t.h[1] = b;
    return t.u;
}
__device__ inline void st32(__bf16* p, unsigned v) { *(unsigned*)(void*)p = v; }
__device__ inline void st128(__bf16* p, u32x4 v)   { *(u32x4*)(void*)p = v; }

// A[m][k] tile, row-major (k contiguous), ldk multiple of 8, base 16B aligned.
__device__ inline v16bf load_a_frag(const __bf16* base, int ldk) {
    int lane = threadIdx.x & 31;
    const __bf16* p = base + (lane & 15) * ldk + ((lane >> 4) << 3);
    return combine(ld8(p), ld8(p + 16));
}
// B tile stored n-major (k contiguous): baseT[n][k], ldk multiple of 8.
__device__ inline v16bf load_b_frag_t(const __bf16* baseT, int ldk) {
    int lane = threadIdx.x & 31;
    const __bf16* p = baseT + (lane & 15) * ldk + ((lane >> 4) << 3);
    return combine(ld8(p), ld8(p + 16));
}

// ---------------- Kernel 1: bilinear upsample of position biases ----------------
__global__ void k_upsample(const float* __restrict__ an, const float* __restrict__ na,
                           float* __restrict__ an_up, float* __restrict__ na_up) {
    int idx = blockIdx.x * 256 + threadIdx.x;
    int total = 2 * HEADS * AG * NPOS;
    if (idx >= total) return;
    int which = idx / (HEADS * AG * NPOS);
    int r = idx % (HEADS * AG * NPOS);
    int ha = r / NPOS;
    int pos = r % NPOS;
    int y = pos / HW, x = pos % HW;
    const float* src = which ? na : an;
    float* dst = which ? na_up : an_up;
    float sy = (y + 0.5f) * (7.0f / 56.0f) - 0.5f;
    float sx = (x + 0.5f) * (7.0f / 56.0f) - 0.5f;
    float fy = floorf(sy), fx = floorf(sx);
    float wy = sy - fy, wx = sx - fx;
    int y0 = (int)fy, x0 = (int)fx;
    int y0c = min(max(y0, 0), 6), y1c = min(max(y0 + 1, 0), 6);
    int x0c = min(max(x0, 0), 6), x1c = min(max(x0 + 1, 0), 6);
    const float* s = src + (size_t)ha * 49;
    float v00 = s[y0c * 7 + x0c], v01 = s[y0c * 7 + x1c];
    float v10 = s[y1c * 7 + x0c], v11 = s[y1c * 7 + x1c];
    float v = (1.0f - wy) * ((1.0f - wx) * v00 + wx * v01) +
              wy * ((1.0f - wx) * v10 + wx * v11);
    dst[(size_t)ha * NPOS + pos] = v;
}

// ---------------- Kernel 2: QKV GEMM (M=b*n, N=1152, K=384), 64x64 block ----------------
__global__ __launch_bounds__(256) void k_qkv_gemm(
    const float* __restrict__ x, const float* __restrict__ w,
    const float* __restrict__ bias,
    __bf16* __restrict__ qo, __bf16* __restrict__ ko, __bf16* __restrict__ vo) {
    __shared__ __bf16 At[64 * 32];    // [m][k]
    __shared__ __bf16 BtS[64 * 32];   // n-major: [n][k]
    int bm = blockIdx.x;              // M / 64 = 1568
    int bn = blockIdx.y;              // 1152 / 64 = 18
    int tid = threadIdx.x, wv = tid >> 5, lane = tid & 31;
    int wrow = wv >> 1, wcol = wv & 1;
    v8f acc0 = {}, acc1 = {};
    int r = tid & 63;                 // A row / B column
    int kq0 = (tid >> 6) * 8;         // 8 consecutive k per thread
    int rowg = bm * 64 + r;
    int bi = rowg / NPOS;
    int pos = rowg - bi * NPOS;
    for (int kt = 0; kt < 12; ++kt) {
        int k0 = kt * 32;
        {   // A: strided f32 reads (coalesced across threads), pack -> one b128 store
            const float* px = &x[(size_t)bi * CDIM * NPOS + (size_t)(k0 + kq0) * NPOS + pos];
            float f[8];
#pragma unroll
            for (int e = 0; e < 8; ++e) f[e] = px[(size_t)e * NPOS];
            u32x4 pk;
            pk[0] = pk2f(f[0], f[1]); pk[1] = pk2f(f[2], f[3]);
            pk[2] = pk2f(f[4], f[5]); pk[3] = pk2f(f[6], f[7]);
            st128(&At[r * 32 + kq0], pk);
        }
        {   // B: n-major staging, pack -> one b128 store
            const float* pw = &w[(size_t)(k0 + kq0) * N3C + bn * 64 + r];
            float f[8];
#pragma unroll
            for (int e = 0; e < 8; ++e) f[e] = pw[(size_t)e * N3C];
            u32x4 pk;
            pk[0] = pk2f(f[0], f[1]); pk[1] = pk2f(f[2], f[3]);
            pk[2] = pk2f(f[4], f[5]); pk[3] = pk2f(f[6], f[7]);
            st128(&BtS[r * 32 + kq0], pk);
        }
        if (kt + 1 < 12) {   // prefetch next K tile
            __builtin_prefetch(
                &x[(size_t)bi * CDIM * NPOS + (size_t)(k0 + 32 + kq0) * NPOS + pos], 0, 1);
            __builtin_prefetch(&w[(size_t)(k0 + 32 + kq0) * N3C + bn * 64 + r], 0, 1);
        }
        __syncthreads();
        v16bf a  = load_a_frag(&At[wrow * 16 * 32], 32);
        v16bf b0 = load_b_frag_t(&BtS[(wcol * 32) * 32], 32);
        v16bf b1 = load_b_frag_t(&BtS[(wcol * 32 + 16) * 32], 32);
        acc0 = __builtin_amdgcn_wmma_f32_16x16x32_bf16(false, a, false, b0, (short)0, acc0,
                                                       false, false);
        acc1 = __builtin_amdgcn_wmma_f32_16x16x32_bf16(false, a, false, b1, (short)0, acc1,
                                                       false, false);
        __syncthreads();
    }
    int cg0 = bn * 64 + wcol * 32 + (lane & 15);
    int cg1 = cg0 + 16;
    int which = cg0 / CDIM;           // 64-col block never straddles q/k/v boundary
    int ch0 = cg0 - which * CDIM, ch1 = cg1 - which * CDIM;
    __bf16* outp = (which == 0) ? qo : ((which == 1) ? ko : vo);
    float bv0 = bias[cg0], bv1 = bias[cg1];
#pragma unroll
    for (int rr = 0; rr < 8; ++rr) {
        int rowg2 = bm * 64 + wrow * 16 + rr + ((lane >> 4) << 3);
        outp[(size_t)rowg2 * CDIM + ch0] = (__bf16)(acc0[rr] + bv0);
        outp[(size_t)rowg2 * CDIM + ch1] = (__bf16)(acc1[rr] + bv1);
    }
}

// ---------------- Kernel 3: agent pooling (8x8 avg of q) ----------------
__global__ void k_pool(const __bf16* __restrict__ q, __bf16* __restrict__ agent) {
    int idx = blockIdx.x * 256 + threadIdx.x;
    if (idx >= BATCH * AG * CDIM) return;
    int ch = idx % CDIM;
    int t = idx / CDIM;
    int a = t % AG;
    int b = t / AG;
    int py = a / 7, px = a % 7;
    float s = 0.0f;
    for (int dy = 0; dy < 8; ++dy)
        for (int dx = 0; dx < 8; ++dx) {
            int pos = (py * 8 + dy) * HW + px * 8 + dx;
            s += (float)q[((size_t)b * NPOS + pos) * CDIM + ch];
        }
    agent[idx] = (__bf16)(s * (1.0f / 64.0f));
}

// ---------------- Kernel 4: stage-1 agent attention (flash over n) ----------------
__global__ __launch_bounds__(256) void k_stage1(
    const __bf16* __restrict__ agent, const __bf16* __restrict__ kmat,
    const __bf16* __restrict__ vmat, const float* __restrict__ an_up,
    __bf16* __restrict__ agent_v) {
    __shared__ float  S[64 * 64];
    __shared__ __bf16 P[64 * 64];     // [row][k] row-major
    __shared__ __bf16 VtT[32 * 64];   // transposed: [d][pos]
    __shared__ float sm_m[64], sm_l[64], sm_alpha[64];
    int bh = blockIdx.x;
    int bi = bh / HEADS, hi = bh % HEADS;
    int tid = threadIdx.x, wv = tid >> 5, lane = tid & 31;
    int rowt = wv >> 1, half = wv & 1;
    const float scale = 0.17677669529663687f;  // 32^-0.5

    // agent A-fragment (constant across chunks); rows >= 49 padded with zero
    v16bf afrag;
    {
        int m = rowt * 16 + (lane & 15);
        if (m < AG) {
            const __bf16* p =
                agent + ((size_t)bi * AG + m) * CDIM + hi * HD + ((lane >> 4) << 3);
            afrag = combine(ld8(p), ld8(p + 16));
        } else {
            v8bf z = {};
            afrag = combine(z, z);
        }
    }
    if (tid < 64) { sm_m[tid] = -1e30f; sm_l[tid] = 0.0f; }
    __syncthreads();
    v8f acc = {};

    // staging decomposition for the transposed V tile (pair of positions / thread)
    int pp = tid & 31;       // position pair: pos0 = pp*2
    int dg = tid >> 5;       // 0..7 -> 4 d values each
    int d0s = dg * 4;

    for (int cc = 0; cc < NPOS / 64; ++cc) {   // 49 chunks of 64 key positions
        int colbase = cc * 64;
        {   // V chunk -> LDS transposed [d][pos], packed pair stores
            int pos0 = colbase + pp * 2;
            const __bf16* g0 = &vmat[((size_t)bi * NPOS + pos0) * CDIM + hi * HD + d0s];
            v4bf a0 = ld4(g0);
            v4bf a1 = ld4(g0 + CDIM);
#pragma unroll
            for (int e = 0; e < 4; ++e)
                st32(&VtT[(d0s + e) * 64 + pp * 2], pk2b(a0[e], a1[e]));
        }
        // QK^T: each wave computes 2 of the 16 (rowtile x subtile) score tiles
#pragma unroll
        for (int s2 = 0; s2 < 2; ++s2) {
            int st = half * 2 + s2;
            v16bf bfrag;   // K fragment direct from global: two 16B loads per lane
            {
                int pos = colbase + st * 16 + (lane & 15);
                const __bf16* p =
                    kmat + ((size_t)bi * NPOS + pos) * CDIM + hi * HD + ((lane >> 4) << 3);
                bfrag = combine(ld8(p), ld8(p + 16));
            }
            v8f sc = {};
            sc = __builtin_amdgcn_wmma_f32_16x16x32_bf16(false, afrag, false, bfrag,
                                                         (short)0, sc, false, false);
            int coll = st * 16 + (lane & 15);
#pragma unroll
            for (int rr = 0; rr < 8; ++rr) {
                int row = rowt * 16 + rr + ((lane >> 4) << 3);
                float val = (row < AG)
                    ? sc[rr] * scale +
                      an_up[((size_t)hi * AG + row) * NPOS + colbase + coll]
                    : 0.0f;
                S[row * 64 + coll] = val;
            }
        }
        __syncthreads();
        if (tid < 64) {   // online softmax per agent row; packed P stores
            float mold = sm_m[tid];
            float mx = mold;
            for (int j = 0; j < 64; ++j) mx = fmaxf(mx, S[tid * 64 + j]);
            float alpha = __expf(mold - mx);
            float lsum = 0.0f;
            for (int j = 0; j < 64; j += 2) {
                float p0 = __expf(S[tid * 64 + j] - mx);
                float p1 = __expf(S[tid * 64 + j + 1] - mx);
                lsum += p0 + p1;
                st32(&P[tid * 64 + j], pk2f(p0, p1));
            }
            sm_m[tid] = mx;
            sm_l[tid] = sm_l[tid] * alpha + lsum;
            sm_alpha[tid] = alpha;
        }
        __syncthreads();
        // rescale running accumulator, then P @ V
#pragma unroll
        for (int rr = 0; rr < 8; ++rr) {
            int row = rowt * 16 + rr + ((lane >> 4) << 3);
            acc[rr] *= sm_alpha[row];
        }
#pragma unroll
        for (int kk = 0; kk < 2; ++kk) {
            v16bf pa = load_a_frag(&P[rowt * 16 * 64 + kk * 32], 64);
            v16bf vb = load_b_frag_t(&VtT[(half * 16) * 64 + kk * 32], 64);
            acc = __builtin_amdgcn_wmma_f32_16x16x32_bf16(false, pa, false, vb, (short)0,
                                                          acc, false, false);
        }
        __syncthreads();
    }
    // normalize and store padded agent_v [bh][64][32]
#pragma unroll
    for (int rr = 0; rr < 8; ++rr) {
        int row = rowt * 16 + rr + ((lane >> 4) << 3);
        int d = half * 16 + (lane & 15);
        float val = (row < AG) ? acc[rr] / sm_l[row] : 0.0f;
        agent_v[(((size_t)bh) * 64 + row) * 32 + d] = (__bf16)val;
    }
}

// ---------------- Kernel 5: stage-2 attention + fused 3x3 depthwise conv ----------------
__global__ __launch_bounds__(256) void k_stage2(
    const __bf16* __restrict__ q, const __bf16* __restrict__ agent,
    const __bf16* __restrict__ agent_v, const float* __restrict__ na_up,
    const __bf16* __restrict__ vmat, const float* __restrict__ dwc_w,
    const float* __restrict__ dwc_b, __bf16* __restrict__ o) {
    __shared__ __bf16 Ag[64 * 32];    // [a][d] (d contiguous), zero padded
    __shared__ __bf16 AvT[32 * 64];   // transposed: [d][a]
    __shared__ float  S[64 * 64];
    __shared__ __bf16 P[64 * 64];
    __shared__ float sm_l[64];
    int idx = blockIdx.x;
    int bh = idx / (NPOS / 64), rt = idx % (NPOS / 64);
    int bi = bh / HEADS, hi = bh % HEADS;
    int rowbase = rt * 64;
    int tid = threadIdx.x, wv = tid >> 5, lane = tid & 31;
    const float scale = 0.17677669529663687f;
    {   // AvT: pair-of-rows transposed staging with b32 packed stores
        int pp = tid & 31;        // a0 = pp*2
        int dg = tid >> 5;        // d0 = dg*4
        int a0 = pp * 2, d0 = dg * 4;
        const __bf16* g0 = &agent_v[(((size_t)bh) * 64 + a0) * 32 + d0];
        v4bf r0 = ld4(g0);
        v4bf r1 = ld4(g0 + 32);
#pragma unroll
        for (int e = 0; e < 4; ++e)
            st32(&AvT[(d0 + e) * 64 + a0], pk2b(r0[e], r1[e]));
    }
    {   // Ag: straight 16B copy, zero pad rows >= AG
        int a = tid >> 2;
        int d0 = (tid & 3) * 8;
        if (a < AG) {
            u32x4 t = *(const u32x4*)(const void*)
                &agent[((size_t)bi * AG + a) * CDIM + hi * HD + d0];
            st128(&Ag[a * 32 + d0], t);
        } else {
            u32x4 z = {};
            st128(&Ag[a * 32 + d0], z);
        }
    }
    __syncthreads();
    int rowt = wv >> 1, chalf = wv & 1;
    v16bf afrag;   // q rows, direct from global (two 16B loads per lane)
    {
        int m = rowbase + rowt * 16 + (lane & 15);
        const __bf16* p =
            q + ((size_t)bi * NPOS + m) * CDIM + hi * HD + ((lane >> 4) << 3);
        afrag = combine(ld8(p), ld8(p + 16));
    }
#pragma unroll
    for (int s2 = 0; s2 < 2; ++s2) {
        int ct = chalf * 2 + s2;
        v16bf bfrag = load_b_frag_t(&Ag[(ct * 16) * 32], 32);
        v8f sc = {};
        sc = __builtin_amdgcn_wmma_f32_16x16x32_bf16(false, afrag, false, bfrag, (short)0,
                                                     sc, false, false);
        int a = ct * 16 + (lane & 15);
#pragma unroll
        for (int rr = 0; rr < 8; ++rr) {
            int row = rowt * 16 + rr + ((lane >> 4) << 3);
            int pos = rowbase + row;
            float val = (a < AG)
                ? sc[rr] * scale + na_up[((size_t)hi * AG + a) * NPOS + pos]
                : -1e30f;
            S[row * 64 + a] = val;
        }
    }
    __syncthreads();
    if (tid < 64) {   // softmax over 49 agents; packed P stores
        float mx = -1e30f;
        for (int j = 0; j < AG; ++j) mx = fmaxf(mx, S[tid * 64 + j]);
        float lsum = 0.0f;
        for (int j = 0; j < 64; j += 2) {
            float p0 = (j     < AG) ? __expf(S[tid * 64 + j] - mx) : 0.0f;
            float p1 = (j + 1 < AG) ? __expf(S[tid * 64 + j + 1] - mx) : 0.0f;
            lsum += p0 + p1;
            st32(&P[tid * 64 + j], pk2f(p0, p1));
        }
        sm_l[tid] = lsum;
    }
    __syncthreads();
    int dt = chalf;
    v8f acc = {};
#pragma unroll
    for (int kk = 0; kk < 2; ++kk) {
        v16bf pa = load_a_frag(&P[rowt * 16 * 64 + kk * 32], 64);
        v16bf vb = load_b_frag_t(&AvT[(dt * 16) * 64 + kk * 32], 64);
        acc = __builtin_amdgcn_wmma_f32_16x16x32_bf16(false, pa, false, vb, (short)0, acc,
                                                      false, false);
    }
    // epilogue: normalize, add depthwise 3x3 conv of v, store bf16
#pragma unroll
    for (int rr = 0; rr < 8; ++rr) {
        int row = rowt * 16 + rr + ((lane >> 4) << 3);
        int pos = rowbase + row;
        int d = dt * 16 + (lane & 15);
        int chq = hi * HD + d;
        float val = acc[rr] / sm_l[row];
        int y = pos / HW, x2 = pos % HW;
        float dw = dwc_b[chq];
#pragma unroll
        for (int dy = -1; dy <= 1; ++dy) {
            int yy = y + dy;
            if (yy < 0 || yy >= HW) continue;
#pragma unroll
            for (int dx = -1; dx <= 1; ++dx) {
                int xx = x2 + dx;
                if (xx < 0 || xx >= HW) continue;
                dw += (float)vmat[((size_t)bi * NPOS + yy * HW + xx) * CDIM + chq] *
                      dwc_w[chq * 9 + (dy + 1) * 3 + (dx + 1)];
            }
        }
        o[((size_t)bi * NPOS + pos) * CDIM + chq] = (__bf16)(val + dw);
    }
}

// ---------------- Kernel 6: proj GEMM + sigmoid gate (NCHW output) ----------------
__global__ __launch_bounds__(256) void k_proj(
    const __bf16* __restrict__ o, const float* __restrict__ w,
    const float* __restrict__ bias, const float* __restrict__ x,
    float* __restrict__ out) {
    __shared__ __bf16 At[64 * 32];    // [m][k]
    __shared__ __bf16 BtS[64 * 32];   // n-major: [n][k]
    int bm = blockIdx.x, bn = blockIdx.y;
    int tid = threadIdx.x, wv = tid >> 5, lane = tid & 31;
    int wrow = wv >> 1, wcol = wv & 1;
    v8f acc0 = {}, acc1 = {};
    int r = tid & 63;
    int kq0 = (tid >> 6) * 8;
    int rowg = bm * 64 + r;
    for (int kt = 0; kt < 12; ++kt) {
        int k0 = kt * 32;
        {   // A from o (bf16): one b128 load + one b128 store
            u32x4 t = *(const u32x4*)(const void*)&o[(size_t)rowg * CDIM + k0 + kq0];
            st128(&At[r * 32 + kq0], t);
        }
        {   // B: pack f32 -> one b128 store
            const float* pw = &w[(size_t)(k0 + kq0) * CDIM + bn * 64 + r];
            float f[8];
#pragma unroll
            for (int e = 0; e < 8; ++e) f[e] = pw[(size_t)e * CDIM];
            u32x4 pk;
            pk[0] = pk2f(f[0], f[1]); pk[1] = pk2f(f[2], f[3]);
            pk[2] = pk2f(f[4], f[5]); pk[3] = pk2f(f[6], f[7]);
            st128(&BtS[r * 32 + kq0], pk);
        }
        if (kt + 1 < 12) {
            __builtin_prefetch(&o[(size_t)rowg * CDIM + k0 + 32 + kq0], 0, 1);
            __builtin_prefetch(&w[(size_t)(k0 + 32 + kq0) * CDIM + bn * 64 + r], 0, 1);
        }
        __syncthreads();
        v16bf a  = load_a_frag(&At[wrow * 16 * 32], 32);
        v16bf b0 = load_b_frag_t(&BtS[(wcol * 32) * 32], 32);
        v16bf b1 = load_b_frag_t(&BtS[(wcol * 32 + 16) * 32], 32);
        acc0 = __builtin_amdgcn_wmma_f32_16x16x32_bf16(false, a, false, b0, (short)0, acc0,
                                                       false, false);
        acc1 = __builtin_amdgcn_wmma_f32_16x16x32_bf16(false, a, false, b1, (short)0, acc1,
                                                       false, false);
        __syncthreads();
    }
    int cg0 = bn * 64 + wcol * 32 + (lane & 15);
    int cg1 = cg0 + 16;
    float bv0 = bias[cg0], bv1 = bias[cg1];
#pragma unroll
    for (int rr = 0; rr < 8; ++rr) {
        int rowg2 = bm * 64 + wrow * 16 + rr + ((lane >> 4) << 3);
        int bi = rowg2 / NPOS;
        int pos = rowg2 - bi * NPOS;
        size_t xi0 = (size_t)bi * CDIM * NPOS + (size_t)cg0 * NPOS + pos;
        size_t xi1 = (size_t)bi * CDIM * NPOS + (size_t)cg1 * NPOS + pos;
        float v0 = acc0[rr] + bv0;
        float v1 = acc1[rr] + bv1;
        out[xi0] = x[xi0] * (1.0f / (1.0f + __expf(-v0)));
        out[xi1] = x[xi1] * (1.0f / (1.0f + __expf(-v1)));
    }
}

extern "C" void kernel_launch(void* const* d_in, const int* in_sizes, int n_in,
                              void* d_out, int out_size, void* d_ws, size_t ws_size,
                              hipStream_t stream) {
    (void)in_sizes; (void)n_in; (void)out_size; (void)ws_size;
    const float* x      = (const float*)d_in[0];
    const float* qkv_w  = (const float*)d_in[1];
    const float* qkv_b  = (const float*)d_in[2];
    const float* proj_w = (const float*)d_in[3];
    const float* proj_b = (const float*)d_in[4];
    const float* an_b   = (const float*)d_in[5];
    const float* na_b   = (const float*)d_in[6];
    const float* dwc_w  = (const float*)d_in[7];
    const float* dwc_b  = (const float*)d_in[8];
    float* out = (float*)d_out;

    char* ws = (char*)d_ws;
    size_t off = 0;
    auto alloc = [&](size_t bytes) -> void* {
        void* p = ws + off;
        off += (bytes + 255) & ~(size_t)255;
        return p;
    };
    const size_t MN = (size_t)BATCH * NPOS;     // 100352 rows
    __bf16* q_bf   = (__bf16*)alloc(MN * CDIM * 2);
    __bf16* k_bf   = (__bf16*)alloc(MN * CDIM * 2);
    __bf16* v_bf   = (__bf16*)alloc(MN * CDIM * 2);
    __bf16* o_bf   = (__bf16*)alloc(MN * CDIM * 2);
    __bf16* agent  = (__bf16*)alloc((size_t)BATCH * AG * CDIM * 2);
    __bf16* ag_v   = (__bf16*)alloc((size_t)BATCH * HEADS * 64 * 32 * 2);
    float*  an_up  = (float*)alloc((size_t)HEADS * AG * NPOS * 4);
    float*  na_up  = (float*)alloc((size_t)HEADS * AG * NPOS * 4);

    {
        int total = 2 * HEADS * AG * NPOS;
        k_upsample<<<(total + 255) / 256, 256, 0, stream>>>(an_b, na_b, an_up, na_up);
    }
    k_qkv_gemm<<<dim3((int)(MN / 64), N3C / 64), 256, 0, stream>>>(
        x, qkv_w, qkv_b, q_bf, k_bf, v_bf);
    {
        int total = BATCH * AG * CDIM;
        k_pool<<<(total + 255) / 256, 256, 0, stream>>>(q_bf, agent);
    }
    k_stage1<<<BATCH * HEADS, 256, 0, stream>>>(agent, k_bf, v_bf, an_up, ag_v);
    k_stage2<<<BATCH * HEADS * (NPOS / 64), 256, 0, stream>>>(
        q_bf, agent, ag_v, na_up, v_bf, dwc_w, dwc_b, o_bf);
    k_proj<<<dim3((int)(MN / 64), CDIM / 64), 256, 0, stream>>>(
        o_bf, proj_w, proj_b, x, out);
}
